// KeyFormer_47330539602584
// MI455X (gfx1250) — compile-verified, matrix operand order checked by reading, not run on hardware
//
#include <hip/hip_runtime.h>

// ---------------------------------------------------------------------------
// KeyFormer block for MI455X (gfx1250, wave32, WMMA + TDM).
// GEMMs run through v_wmma_f32_16x16x32_bf16; the scores GEMM stages its
// (pure-copy) bf16 tiles with a double-buffered TENSOR_LOAD_TO_LDS pipeline.
// The s_wait_tensorcnt is inline asm carrying the LDS tile pointers as
// operands + memory clobber, so the compiler knows the DMA wrote the tiles
// (otherwise the ds_load -> wmma chain folds to undef, as seen in round 2).
// ---------------------------------------------------------------------------

typedef __attribute__((ext_vector_type(16))) __bf16 v16bf;
typedef __attribute__((ext_vector_type(8)))  float  v8f;
typedef __attribute__((ext_vector_type(4)))  unsigned int u32x4;
typedef __attribute__((ext_vector_type(8)))  int          i32x8;
typedef __attribute__((ext_vector_type(4)))  int          i32x4;

#define M_DIM 512
#define N_DIM 4096
#define D_DIM 128
#define L_DIM 2048
#define H_DIM 32

// ---- helpers ---------------------------------------------------------------

__device__ __forceinline__ unsigned int f2bf1(float f) {
  unsigned int u = __float_as_uint(f);
  return (u + 0x7FFFu + ((u >> 16) & 1u)) >> 16;   // round-to-nearest-even
}
// RNE-round two f32 and merge the bf16 pair with a single v_perm_b32.
__device__ __forceinline__ unsigned int pack2(float lo, float hi) {
  unsigned int ul = __float_as_uint(lo);
  unsigned int uh = __float_as_uint(hi);
  ul += 0x7FFFu + ((ul >> 16) & 1u);
  uh += 0x7FFFu + ((uh >> 16) & 1u);
  return __builtin_amdgcn_perm(uh, ul, 0x07060302u);  // {uh[3],uh[2],ul[3],ul[2]}
}

struct FragU { union { uint4 u[2]; v16bf v; }; };

// ---- TDM: 2D bf16 tile (tile_rows x tile_elems) global -> LDS --------------
// D# per CDNA5 ISA 8.3/8.4: group0 = {count/flags, lds_addr, global_addr, type},
// group1 = {data_size, tensor_dim0/1, tile_dim0/1/2, tensor_dim0_stride}.
// Rows land contiguously in LDS: row r at lds_off + r*tile_elems*2 bytes.

#if __has_builtin(__builtin_amdgcn_tensor_load_to_lds)
#define HAVE_TDM 1
__device__ __forceinline__ void tdm_load_2d_bf16(
    const unsigned short* gsrc, unsigned int lds_off,
    unsigned int rows_total,   // tensor_dim1 (rows in tensor)
    unsigned int row_elems,    // tensor_dim0 and dim0 stride (elements)
    unsigned int tile_rows,    // tile_dim1
    unsigned int tile_elems) { // tile_dim0
  unsigned long long ga = (unsigned long long)(size_t)gsrc;
  u32x4 g0;
  g0[0] = 1u;                                       // count=1, user descriptor
  g0[1] = lds_off;                                  // lds_addr (bytes)
  g0[2] = (unsigned int)(ga & 0xFFFFFFFFu);         // global_addr[31:0]
  g0[3] = (unsigned int)((ga >> 32) & 0x01FFFFFFu)  // global_addr[56:32]
          | (2u << 30);                             // type = 2 ("image")
  i32x8 g1;
  g1[0] = (int)(1u << 16);                          // data_size = 1 -> 2 bytes
  g1[1] = (int)(row_elems << 16);                   // tensor_dim0[15:0]
  g1[2] = (int)((row_elems >> 16) | (rows_total << 16)); // td0[31:16] | td1[15:0]
  g1[3] = (int)((rows_total >> 16) | (tile_elems << 16)); // td1[31:16] | tile_dim0
  g1[4] = (int)(tile_rows & 0xFFFFu);               // tile_dim1 (tile_dim2 = 0)
  g1[5] = (int)row_elems;                           // tensor_dim0_stride[31:0]
  g1[6] = 0;                                        // stride[47:32] | td1_stride lo
  g1[7] = 0;
  i32x4 z4 = {0, 0, 0, 0};
#if __clang_major__ >= 23
  i32x8 z8 = {0, 0, 0, 0, 0, 0, 0, 0};
  __builtin_amdgcn_tensor_load_to_lds(g0, g1, z4, z4, z8, 0);
#else
  __builtin_amdgcn_tensor_load_to_lds(g0, g1, z4, z4, 0);
#endif
}

// Wait for TENSORcnt and tell the compiler the DMA wrote the LDS tiles:
// the tile base pointers escape into the asm, and the memory clobber makes
// every later ds_load actually read LDS instead of folding to undef.
__device__ __forceinline__ void tdm_wait2_fence(const void* a, const void* b) {
  asm volatile("s_wait_tensorcnt 0x2" :: "s"(a), "s"(b) : "memory");
}
__device__ __forceinline__ void tdm_wait0_fence(const void* a, const void* b) {
  asm volatile("s_wait_tensorcnt 0x0" :: "s"(a), "s"(b) : "memory");
}
#else
#define HAVE_TDM 0
#endif

// ---- LDS tile loaders (tiles stored as [row][k-pair], 16 dwords per row) ---

// A tile 128(rows) x 32(k), f32 source (row-major, ld = lda), convert->bf16
__device__ __forceinline__ void load_tileA_f32(const float* __restrict__ A, int lda,
                                               unsigned int* sT) {
  int t = threadIdx.x; int m = t >> 1; int kh = t & 1;
  const float4* p = (const float4*)(A + (size_t)m * lda + kh * 16);
  float4 f0 = p[0], f1 = p[1], f2 = p[2], f3 = p[3];
  uint4* s4 = (uint4*)(sT + m * 16 + kh * 8);
  s4[0] = make_uint4(pack2(f0.x, f0.y), pack2(f0.z, f0.w),
                     pack2(f1.x, f1.y), pack2(f1.z, f1.w));
  s4[1] = make_uint4(pack2(f2.x, f2.y), pack2(f2.z, f2.w),
                     pack2(f3.x, f3.y), pack2(f3.z, f3.w));
}

// A tile 128 x 32, bf16 source (row-major contiguous k) — fallback copy path
__device__ __forceinline__ void load_tileA_bf16(const unsigned short* __restrict__ A, int lda,
                                                unsigned int* sT) {
  int t = threadIdx.x; int m = t >> 1; int kh = t & 1;
  const uint4* p = (const uint4*)(A + (size_t)m * lda + kh * 16);
  uint4* s4 = (uint4*)(sT + m * 16 + kh * 8);
  s4[0] = p[0];
  s4[1] = p[1];
}

// B tile from row-major W[k][n] (32 x 128), transpose -> LDS Bt[n][k-pair]
__device__ __forceinline__ void load_tileB_wT(const float* __restrict__ B, int ldb,
                                              unsigned int* sT) {
  int t = threadIdx.x; int kp = t >> 4; int n0 = (t & 15) * 8; int k = kp * 2;
  const float4* p0 = (const float4*)(B + (size_t)k * ldb + n0);
  const float4* p1 = (const float4*)(B + (size_t)(k + 1) * ldb + n0);
  float4 a0 = p0[0], a1 = p0[1], b0 = p1[0], b1 = p1[1];
  float va[8] = {a0.x, a0.y, a0.z, a0.w, a1.x, a1.y, a1.z, a1.w};
  float vb[8] = {b0.x, b0.y, b0.z, b0.w, b1.x, b1.y, b1.z, b1.w};
#pragma unroll
  for (int j = 0; j < 8; j++) sT[(n0 + j) * 16 + kp] = pack2(va[j], vb[j]);
}

// B tile from bf16 V[l][d] (32 x 128), transpose -> LDS Bt[d][l-pair]
__device__ __forceinline__ void load_tileB_vT(const unsigned short* __restrict__ B, int ldb,
                                              unsigned int* sT) {
  int t = threadIdx.x; int lp = t >> 4; int d0 = (t & 15) * 8; int l = lp * 2;
  uint4 a = *(const uint4*)(B + (size_t)l * ldb + d0);
  uint4 b = *(const uint4*)(B + (size_t)(l + 1) * ldb + d0);
  const unsigned short* ua = (const unsigned short*)&a;
  const unsigned short* ub = (const unsigned short*)&b;
#pragma unroll
  for (int j = 0; j < 8; j++)
    sT[(d0 + j) * 16 + lp] = (unsigned int)ua[j] | ((unsigned int)ub[j] << 16);
}

// A tile 128 x 32 from f32 scores with fused softmax weight exp(s-max)/sum
__device__ __forceinline__ void load_tileA_softmax(const float* __restrict__ A, int lda,
                                                   const float4* __restrict__ stats,
                                                   unsigned int* sT) {
  int t = threadIdx.x; int m = t >> 1; int kh = t & 1;
  float4 st = stats[m];
  const float4* p = (const float4*)(A + (size_t)m * lda + kh * 16);
  float4 f[4]; f[0] = p[0]; f[1] = p[1]; f[2] = p[2]; f[3] = p[3];
  float w[16];
#pragma unroll
  for (int j = 0; j < 4; j++) {
    w[4 * j + 0] = __expf(f[j].x - st.x) * st.y;
    w[4 * j + 1] = __expf(f[j].y - st.x) * st.y;
    w[4 * j + 2] = __expf(f[j].z - st.x) * st.y;
    w[4 * j + 3] = __expf(f[j].w - st.x) * st.y;
  }
  uint4* s4 = (uint4*)(sT + m * 16 + kh * 8);
  s4[0] = make_uint4(pack2(w[0], w[1]), pack2(w[2], w[3]),
                     pack2(w[4], w[5]), pack2(w[6], w[7]));
  s4[1] = make_uint4(pack2(w[8], w[9]), pack2(w[10], w[11]),
                     pack2(w[12], w[13]), pack2(w[14], w[15]));
}

// ---- wave WMMA step: 8 waves, wave tile 64(M) x 32(N), 8 wmma per K=32 -----

__device__ __forceinline__ void wmma_step(const unsigned int* sA, const unsigned int* sB,
                                          v8f c[4][2]) {
  int lane = threadIdx.x & 31;
  int wave = threadIdx.x >> 5;
  int wm = wave >> 2, wn = wave & 3;
  int lrow = lane & 15, sel = lane >> 4;
  const uint4* A4 = (const uint4*)sA;
  const uint4* B4 = (const uint4*)sB;
  FragU a[4], b[2];
#pragma unroll
  for (int mi = 0; mi < 4; mi++) {
    int row = wm * 64 + mi * 16 + lrow;
    a[mi].u[0] = A4[row * 4 + sel];
    a[mi].u[1] = A4[row * 4 + 2 + sel];
  }
#pragma unroll
  for (int ni = 0; ni < 2; ni++) {
    int row = wn * 32 + ni * 16 + lrow;
    b[ni].u[0] = B4[row * 4 + sel];
    b[ni].u[1] = B4[row * 4 + 2 + sel];
  }
#pragma unroll
  for (int mi = 0; mi < 4; mi++)
#pragma unroll
    for (int ni = 0; ni < 2; ni++)
      c[mi][ni] = __builtin_amdgcn_wmma_f32_16x16x32_bf16(
          false, a[mi].v, false, b[ni].v, (short)0, c[mi][ni], false, false);
}

// ---- kernel 0: convert caches f32 -> bf16 (the "updated cache" copies) -----

__global__ __launch_bounds__(256) void cvt_cache_kernel(
    const float* __restrict__ cK, const float* __restrict__ cV,
    unsigned short* __restrict__ kbf, unsigned short* __restrict__ vbf) {
  size_t i4 = (size_t)blockIdx.x * 256 + threadIdx.x;   // float4 index
  float4 a = ((const float4*)cK)[i4];
  float4 b = ((const float4*)cV)[i4];
  ((uint2*)kbf)[i4] = make_uint2(pack2(a.x, a.y), pack2(a.z, a.w));
  ((uint2*)vbf)[i4] = make_uint2(pack2(b.x, b.y), pack2(b.z, b.w));
}

// ---- kernel 1: fused QKV GEMM, epilogue scatters to bf16 q / cache -------

__global__ __launch_bounds__(256) void qkv_gemm_kernel(
    const float* __restrict__ X,
    const float* __restrict__ Wq, const float* __restrict__ Wk, const float* __restrict__ Wv,
    const int* __restrict__ Pp,
    unsigned short* __restrict__ qbf, unsigned short* __restrict__ kbf,
    unsigned short* __restrict__ vbf) {
  __shared__ unsigned int sA[128 * 16];
  __shared__ unsigned int sB[128 * 16];
  int bx = blockIdx.x;                 // 0..95 over 3N
  int m0 = blockIdx.y * 128;
  int wsel = bx >> 5;                  // 0=q, 1=k, 2=v
  int n0 = (bx & 31) * 128;
  const float* W = (wsel == 0) ? Wq : ((wsel == 1) ? Wk : Wv);
  v8f c[4][2] = {};
  for (int k0 = 0; k0 < N_DIM; k0 += 32) {
    load_tileA_f32(X + (size_t)m0 * N_DIM + k0, N_DIM, sA);
    load_tileB_wT(W + (size_t)k0 * N_DIM + n0, N_DIM, sB);
    if (k0 + 32 < N_DIM) {
      __builtin_prefetch(X + (size_t)m0 * N_DIM + k0 + 32 + (threadIdx.x << 4), 0, 0);
      __builtin_prefetch(W + (size_t)(k0 + 32) * N_DIM + n0 + (threadIdx.x >> 1), 0, 0);
    }
    __syncthreads();
    wmma_step(sA, sB, c);
    __syncthreads();
  }
  int Pv = *Pp;
  int lane = threadIdx.x & 31, wave = threadIdx.x >> 5;
  int wm = wave >> 2, wn = wave & 3, sel = lane >> 4;
#pragma unroll
  for (int mi = 0; mi < 4; mi++)
#pragma unroll
    for (int ni = 0; ni < 2; ni++)
#pragma unroll
      for (int r = 0; r < 8; r++) {
        int gm = m0 + wm * 64 + mi * 16 + 8 * sel + r;
        int gn = n0 + wn * 32 + ni * 16 + (lane & 15);
        int h = gn >> 7, d = gn & 127;
        unsigned short bv = (unsigned short)f2bf1(c[mi][ni][r]);
        if (wsel == 0)
          qbf[(size_t)h * M_DIM * D_DIM + (size_t)gm * D_DIM + d] = bv;
        else if (wsel == 1)
          kbf[(size_t)h * L_DIM * D_DIM + (size_t)(Pv + gm) * D_DIM + d] = bv;
        else
          vbf[(size_t)h * L_DIM * D_DIM + (size_t)(Pv + gm) * D_DIM + d] = bv;
      }
}

// ---- kernel 2: scores = q @ K^T; double-buffered TDM tile pipeline --------

__global__ __launch_bounds__(256) void scores_gemm_kernel(
    const unsigned short* __restrict__ qbf, const unsigned short* __restrict__ kbf,
    float* __restrict__ Sout) {
  __shared__ unsigned int sA[2][128 * 16];  // LDS 0 / 8192
  __shared__ unsigned int sB[2][128 * 16];  // LDS 16384 / 24576
  int l0 = blockIdx.x * 128, m0 = blockIdx.y * 128, h = blockIdx.z;
  const unsigned short* Aq = qbf + (size_t)h * M_DIM * D_DIM;
  const unsigned short* Bk = kbf + (size_t)h * L_DIM * D_DIM;
  v8f c[4][2] = {};
#if HAVE_TDM
  const int NIT = D_DIM / 32;                       // 4 K-steps
  bool w0 = (threadIdx.x >> 5) == 0;                // wave-uniform DMA issuer
  if (w0) {
    tdm_load_2d_bf16(Aq + (size_t)m0 * D_DIM, 0u,     M_DIM, D_DIM, 128, 32);
    tdm_load_2d_bf16(Bk + (size_t)l0 * D_DIM, 16384u, L_DIM, D_DIM, 128, 32);
  }
#pragma unroll
  for (int i = 0; i < NIT; ++i) {
    if (w0) {
      if (i + 1 < NIT) {   // prefetch next K-tile into the other buffer
        int kn = (i + 1) * 32;
        unsigned int buf = (unsigned int)((i + 1) & 1) * 8192u;
        tdm_load_2d_bf16(Aq + (size_t)m0 * D_DIM + kn, buf,          M_DIM, D_DIM, 128, 32);
        tdm_load_2d_bf16(Bk + (size_t)l0 * D_DIM + kn, 16384u + buf, L_DIM, D_DIM, 128, 32);
        tdm_wait2_fence(sA, sB);   // current buffer's 2 DMAs done (in-order)
      } else {
        tdm_wait0_fence(sA, sB);
      }
    }
    __syncthreads();
    wmma_step(sA[i & 1], sB[i & 1], c);
    __syncthreads();
  }
#else
  for (int k0 = 0; k0 < D_DIM; k0 += 32) {
    load_tileA_bf16(Aq + (size_t)m0 * D_DIM + k0, D_DIM, sA[0]);
    load_tileA_bf16(Bk + (size_t)l0 * D_DIM + k0, D_DIM, sB[0]);  // same layout
    __syncthreads();
    wmma_step(sA[0], sB[0], c);
    __syncthreads();
  }
#endif
  float* S = Sout + (size_t)h * M_DIM * L_DIM;
  int lane = threadIdx.x & 31, wave = threadIdx.x >> 5;
  int wm = wave >> 2, wn = wave & 3, sel = lane >> 4;
#pragma unroll
  for (int mi = 0; mi < 4; mi++)
#pragma unroll
    for (int ni = 0; ni < 2; ni++)
#pragma unroll
      for (int r = 0; r < 8; r++) {
        int gm = m0 + wm * 64 + mi * 16 + 8 * sel + r;
        int gl = l0 + wn * 32 + ni * 16 + (lane & 15);
        S[(size_t)gm * L_DIM + gl] = c[mi][ni][r];
      }
}

// ---- kernel 3: per-row stats for both softmaxes ---------------------------

__global__ __launch_bounds__(256) void rowstats_kernel(
    const float* __restrict__ S, const float* __restrict__ noise,
    float* __restrict__ stats) {
  int row = blockIdx.x;                        // h*512 + m
  const float* s = S + (size_t)row * L_DIM;
  const float* nz = noise + (size_t)row * L_DIM;
  int t = threadIdx.x;
  float vs[8], vt[8];
  float mx1 = -3.4e38f, mx2 = -3.4e38f;
#pragma unroll
  for (int i = 0; i < 8; i++) {
    float a = s[t + i * 256];
    float b = (a + nz[t + i * 256]) * (1.0f / 1.5f);
    vs[i] = a; vt[i] = b;
    mx1 = fmaxf(mx1, a); mx2 = fmaxf(mx2, b);
  }
  __shared__ float r1[256], r2[256];
  r1[t] = mx1; r2[t] = mx2; __syncthreads();
  for (int off = 128; off; off >>= 1) {
    if (t < off) { r1[t] = fmaxf(r1[t], r1[t + off]); r2[t] = fmaxf(r2[t], r2[t + off]); }
    __syncthreads();
  }
  mx1 = r1[0]; mx2 = r2[0]; __syncthreads();
  float s1 = 0.f, s2 = 0.f;
#pragma unroll
  for (int i = 0; i < 8; i++) { s1 += __expf(vs[i] - mx1); s2 += __expf(vt[i] - mx2); }
  r1[t] = s1; r2[t] = s2; __syncthreads();
  for (int off = 128; off; off >>= 1) {
    if (t < off) { r1[t] += r1[t + off]; r2[t] += r2[t + off]; }
    __syncthreads();
  }
  if (t == 0) {
    float4 st; st.x = mx1; st.y = 1.0f / r1[0]; st.z = mx2; st.w = 1.0f / r2[0];
    ((float4*)stats)[row] = st;
  }
}

// ---- kernel 4: out = softmax(S) @ V (softmax fused into A-fragment build) --

__global__ __launch_bounds__(256) void attnv_gemm_kernel(
    const float* __restrict__ S, const unsigned short* __restrict__ vbf,
    const float4* __restrict__ stats, float* __restrict__ out0) {
  __shared__ unsigned int sA[128 * 16];
  __shared__ unsigned int sB[128 * 16];
  int m0 = blockIdx.x * 128, h = blockIdx.y;
  const float* Sh = S + (size_t)h * M_DIM * L_DIM + (size_t)m0 * L_DIM;
  const unsigned short* Vh = vbf + (size_t)h * L_DIM * D_DIM;
  const float4* sth = stats + (size_t)h * M_DIM + m0;
  v8f c[4][2] = {};
  for (int k0 = 0; k0 < L_DIM; k0 += 32) {
    load_tileA_softmax(Sh + k0, L_DIM, sth, sA);
    load_tileB_vT(Vh + (size_t)k0 * D_DIM, D_DIM, sB);
    if (k0 + 32 < L_DIM)
      __builtin_prefetch(Sh + k0 + 32 + ((threadIdx.x >> 1) * L_DIM), 0, 0);
    __syncthreads();
    wmma_step(sA, sB, c);
    __syncthreads();
  }
  int lane = threadIdx.x & 31, wave = threadIdx.x >> 5;
  int wm = wave >> 2, wn = wave & 3, sel = lane >> 4;
#pragma unroll
  for (int mi = 0; mi < 4; mi++)
#pragma unroll
    for (int ni = 0; ni < 2; ni++)
#pragma unroll
      for (int r = 0; r < 8; r++) {
        int gm = m0 + wm * 64 + mi * 16 + 8 * sel + r;
        int gd = wn * 32 + ni * 16 + (lane & 15);
        out0[(size_t)gm * N_DIM + h * D_DIM + gd] = c[mi][ni][r];
      }
}

// ---- kernel 5: overwrite scores region with perturbed softmax -------------

__global__ __launch_bounds__(256) void perturb_kernel(
    float* __restrict__ S, const float* __restrict__ noise,
    const float4* __restrict__ stats) {
  size_t idx = ((size_t)blockIdx.x * 256 + threadIdx.x) * 4;
  size_t row = idx >> 11;                       // L = 2048
  float4 st = stats[row];
  float4 s = *(const float4*)(S + idx);
  float4 nz = *(const float4*)(noise + idx);
  float4 o;
  o.x = __expf((s.x + nz.x) * (1.0f / 1.5f) - st.z) * st.w;
  o.y = __expf((s.y + nz.y) * (1.0f / 1.5f) - st.z) * st.w;
  o.z = __expf((s.z + nz.z) * (1.0f / 1.5f) - st.z) * st.w;
  o.w = __expf((s.w + nz.w) * (1.0f / 1.5f) - st.z) * st.w;
  *(float4*)(S + idx) = o;
}

// ---------------------------------------------------------------------------

extern "C" void kernel_launch(void* const* d_in, const int* in_sizes, int n_in,
                              void* d_out, int out_size, void* d_ws, size_t ws_size,
                              hipStream_t stream) {
  (void)in_sizes; (void)n_in; (void)out_size; (void)ws_size;
  const float* X     = (const float*)d_in[0];
  const float* Wq    = (const float*)d_in[1];
  const float* Wk    = (const float*)d_in[2];
  const float* Wv    = (const float*)d_in[3];
  const float* noise = (const float*)d_in[4];
  const float* cK    = (const float*)d_in[5];
  const float* cV    = (const float*)d_in[6];
  const int*   Pp    = (const int*)d_in[7];

  char* ws = (char*)d_ws;
  unsigned short* qbf = (unsigned short*)ws;                  //  4 MiB  [H,M,D] bf16
  unsigned short* kbf = (unsigned short*)(ws + 4194304);      // 16 MiB  [H,L,D] bf16
  unsigned short* vbf = (unsigned short*)(ws + 20971520);     // 16 MiB  [H,L,D] bf16
  float*          stats = (float*)(ws + 37748736);            // 256 KiB [H*M,4] f32

  float* out0 = (float*)d_out;                                // [M, N] f32
  float* S    = out0 + (size_t)M_DIM * N_DIM;                 // [H, M, L] f32 (scores -> perturb)

  cvt_cache_kernel<<<8192, 256, 0, stream>>>(cK, cV, kbf, vbf);
  qkv_gemm_kernel<<<dim3(96, 4), 256, 0, stream>>>(X, Wq, Wk, Wv, Pp, qbf, kbf, vbf);
  scores_gemm_kernel<<<dim3(16, 4, 32), 256, 0, stream>>>(qbf, kbf, S);
  rowstats_kernel<<<16384, 256, 0, stream>>>(S, noise, stats);
  attnv_gemm_kernel<<<dim3(4, 32), 256, 0, stream>>>(S, vbf, (const float4*)stats, out0);
  perturb_kernel<<<32768, 256, 0, stream>>>(S, noise, (const float4*)stats);
}
